// CSTreeLSTM_27504970564112
// MI455X (gfx1250) — matrix operand order repaired and checked by reading, not compiled, not run on hardware
//
#include <hip/hip_runtime.h>

// ---------------------------------------------------------------------------
// CDNA5 (gfx1250) Child-Sum TreeLSTM.
// bf16 WMMA (v_wmma_f32_16x16x32_bf16) with fp32 accumulation, wave32.
// Weights staged per-workgroup into LDS with GLOBAL_LOAD_ASYNC_TO_LDS_B128
// (ASYNCcnt) and double buffering; x pre-cast to bf16 so the A stream is
// half-width and conversion-free in the hot loops.
// ---------------------------------------------------------------------------

#define IN_DIM 512
#define HID_DIM 512
#define WROW 1024            // weight row length (IN + HID)
#define WELEM (512 * 1024)   // elements per weight matrix
#define NNODES 69905         // total tree nodes
#define LEAF_OFF 4369        // first leaf row

typedef __attribute__((ext_vector_type(16))) __bf16 v16bf;
typedef __attribute__((ext_vector_type(8)))  float  v8f;

__device__ __forceinline__ float sigmoidf_(float v) {
  return 1.0f / (1.0f + __expf(-v));
}

__device__ __forceinline__ v8f wmma_bf16(v16bf a, v16bf b, v8f c) {
  return __builtin_amdgcn_wmma_f32_16x16x32_bf16(false, a, false, b, (short)0, c,
                                                 false, false);
}

// ---- CDNA5 async copy: global -> LDS, 16B per lane, tracked by ASYNCcnt ----
__device__ __forceinline__ void async_ld16(unsigned lds_off, const void* gaddr) {
  asm volatile("global_load_async_to_lds_b128 %0, %1, off"
               :: "v"(lds_off), "v"(gaddr) : "memory");
}
__device__ __forceinline__ void wait_async3() {
  asm volatile("s_wait_asynccnt 0x3" ::: "memory");
}
__device__ __forceinline__ void wait_async1() {
  asm volatile("s_wait_asynccnt 0x1" ::: "memory");
}
__device__ __forceinline__ void wait_async0() {
  asm volatile("s_wait_asynccnt 0x0" ::: "memory");
}

// A fragment (16x32 MxK). `row` points at this lane's row (m = lane%16).
// Lane layout: elems 0..15 = K[(lane/16)*16 .. +15].
__device__ __forceinline__ v16bf load_a_bf16(const __bf16* __restrict__ row,
                                             int k0, int lane) {
  int sub = (lane >> 4) << 4;
  const __bf16* p0 = row + k0 + sub;
  const __bf16* p1 = p0 + 8;
  v16bf a;
#pragma unroll
  for (int i = 0; i < 8; ++i) { a[i] = p0[i]; a[i + 8] = p1[i]; }
  return a;
}

// Same, but fp32 source (child C in forget): convert on the fly.
__device__ __forceinline__ v16bf load_a_f32(const float* __restrict__ row,
                                            int k0, int lane) {
  int sub = (lane >> 4) << 4;
  const float* p0 = row + k0 + sub;
  const float* p1 = p0 + 8;
  v16bf a;
#pragma unroll
  for (int i = 0; i < 8; ++i) { a[i] = (__bf16)p0[i]; a[i + 8] = (__bf16)p1[i]; }
  return a;
}

// B fragment (32x16 KxN) of W^T straight from global; W row-major [512][1024].
__device__ __forceinline__ v16bf load_b(const __bf16* __restrict__ w, int nb,
                                        int wcol, int k0, int lane) {
  int n = nb + (lane & 15);
  int kk = k0 + ((lane >> 4) << 4);
  return *(const v16bf*)(w + (size_t)n * WROW + wcol + kk);
}

// B fragment from an LDS k-step stage laid out as [gate][64 cols][32 K] bf16.
__device__ __forceinline__ v16bf load_b_lds(const __bf16* base, int g, int t,
                                            int lane) {
  const __bf16* p = base + ((g * 64 + t * 16 + (lane & 15)) * 32 +
                            ((lane >> 4) << 4));
  return *(const v16bf*)p;
}

// ---------------------------------------------------------------------------
// Cast weights (fp32 -> bf16, order i,o,u,f) and ALL x rows to bf16.
// ---------------------------------------------------------------------------
__global__ void k_cast(const float* __restrict__ wi, const float* __restrict__ wo,
                       const float* __restrict__ wu, const float* __restrict__ wf,
                       const float* __restrict__ x,
                       __bf16* __restrict__ wbf, __bf16* __restrict__ xbf) {
  int tid = blockIdx.x * blockDim.x + threadIdx.x;
  int stride = gridDim.x * blockDim.x;
  for (int i = tid; i < WELEM; i += stride) {
    wbf[i]             = (__bf16)wi[i];
    wbf[WELEM + i]     = (__bf16)wo[i];
    wbf[2 * WELEM + i] = (__bf16)wu[i];
    wbf[3 * WELEM + i] = (__bf16)wf[i];
  }
  const int XN = NNODES * IN_DIM;
  for (int i = tid; i < XN; i += stride) xbf[i] = (__bf16)x[i];
}

// ---------------------------------------------------------------------------
// Leaves (level 4, 65536 nodes). Block = 256 threads = 8 waves = 8 parents,
// one colgroup. Wave: its parent's 16 leaf children x 64 cols, gates i,o,u
// fused (h=0 so K=512). Per-k-step 12KB weight tile double-buffered in LDS
// via async-to-LDS; step k's WMMAs overlap step k+1's fetch. Epilogue reduces
// H across the 16 siblings straight into h_sum[parent].
// ---------------------------------------------------------------------------
__global__ __launch_bounds__(256) void k_leaf(
    const __bf16* __restrict__ xbf, const __bf16* __restrict__ wbf,
    const float* __restrict__ bi, const float* __restrict__ bo,
    const float* __restrict__ bu, float* __restrict__ Cleaf,
    __bf16* __restrict__ hsum) {
  __shared__ __bf16 smem[2][3 * 64 * 32];  // 2 x 12KB
  int tid = threadIdx.x;
  int lane = tid & 31;
  int pg = blockIdx.x >> 3;
  int cg = blockIdx.x & 7;
  int p = pg * 8 + (tid >> 5);  // 0..4095
  int nbase = cg * 64;
  const __bf16* arow =
      xbf + (size_t)(LEAF_OFF + p * 16 + (lane & 15)) * IN_DIM;

  auto stage = [&](int buf, int k0) {
#pragma unroll
    for (int it = 0; it < 3; ++it) {
      int c = tid + it * 256;  // 0..767 chunks of 16B
      int g = c >> 8, r = c & 255, n = r >> 2, q = r & 3;
      const __bf16* src =
          wbf + (size_t)g * WELEM + (size_t)(nbase + n) * WROW + k0 + q * 8;
      unsigned loff = (unsigned)(size_t)(&smem[buf][(g * 64 + n) * 32 + q * 8]);
      async_ld16(loff, src);
    }
  };

  stage(0, 0);
  v8f acc[3][4] = {};
  for (int ks = 0; ks < 16; ++ks) {
    int k0 = ks * 32;
    int cur = ks & 1;
    __syncthreads();  // all waves done reading this parity two steps ago
    if (ks + 1 < 16) {
      stage(cur ^ 1, k0 + 32);
      wait_async3();  // our 3 cur-stage asyncs (issued earlier) have landed
    } else {
      wait_async0();
    }
    __syncthreads();  // all waves' shares of buf[cur] visible
    v16bf a = load_a_bf16(arow, k0, lane);
    const __bf16* bb = &smem[cur][0];
#pragma unroll
    for (int g = 0; g < 3; ++g)
#pragma unroll
      for (int t = 0; t < 4; ++t)
        acc[g][t] = wmma_bf16(a, load_b_lds(bb, g, t, lane), acc[g][t]);
  }

  int mhi = (lane >> 4) << 3;
#pragma unroll
  for (int t = 0; t < 4; ++t) {
    int n = nbase + t * 16 + (lane & 15);
    float bbi = bi[n], bbo = bo[n], bbu = bu[n];
    float hpart = 0.f;
#pragma unroll
    for (int j = 0; j < 8; ++j) {
      int m = mhi + j;
      float iv = sigmoidf_(acc[0][t][j] + bbi);
      float ov = sigmoidf_(acc[1][t][j] + bbo);
      float uv = tanhf(acc[2][t][j] + bbu);
      float C = iv * uv;
      float H = ov * tanhf(C);
      Cleaf[(size_t)(p * 16 + m) * HID_DIM + n] = C;
      hpart += H;
    }
    hpart += __shfl_down(hpart, 16, 32);
    if (lane < 16) hsum[(size_t)p * HID_DIM + n] = (__bf16)hpart;
  }
}

// ---------------------------------------------------------------------------
// Internal gates: pre_i/o/u = x@Wx^T + hsum@Wh^T (raw), pre_fx = x@Wfx^T.
// One wave = 16 nodes x 64 cols, 4 gates fused; B fragments gathered into
// registers before the WMMAs so loads clause.
// ---------------------------------------------------------------------------
__global__ __launch_bounds__(128) void k_gates(
    int n, int xoff, const __bf16* __restrict__ xbf,
    const __bf16* __restrict__ hsum, const __bf16* __restrict__ wbf,
    float* __restrict__ pre_i, float* __restrict__ pre_o,
    float* __restrict__ pre_u, float* __restrict__ pre_fx) {
  int lane = threadIdx.x & 31;
  int wave = (blockIdx.x * blockDim.x + threadIdx.x) >> 5;
  int mtiles = (n + 15) >> 4;
  int mt = wave >> 3;
  int cg = wave & 7;
  if (mt >= mtiles) return;
  int nbase = cg * 64;
  int mloc = mt * 16 + (lane & 15);
  int mclamp = mloc < n ? mloc : n - 1;  // pad rows clamp; stores guarded
  const __bf16* ax = xbf + (size_t)(xoff + mclamp) * IN_DIM;
  const __bf16* ah = hsum + (size_t)mclamp * HID_DIM;
  const __bf16* w[4] = {wbf, wbf + WELEM, wbf + 2 * WELEM, wbf + 3 * WELEM};

  v8f acc[4][4] = {};
  for (int k0 = 0; k0 < IN_DIM; k0 += 32) {  // x part: 4 gates, W cols [0,512)
    v16bf a = load_a_bf16(ax, k0, lane);
#pragma unroll
    for (int t = 0; t < 4; ++t) {
      int nb = nbase + t * 16;
      v16bf b0 = load_b(w[0], nb, 0, k0, lane);
      v16bf b1 = load_b(w[1], nb, 0, k0, lane);
      v16bf b2 = load_b(w[2], nb, 0, k0, lane);
      v16bf b3 = load_b(w[3], nb, 0, k0, lane);
      acc[0][t] = wmma_bf16(a, b0, acc[0][t]);
      acc[1][t] = wmma_bf16(a, b1, acc[1][t]);
      acc[2][t] = wmma_bf16(a, b2, acc[2][t]);
      acc[3][t] = wmma_bf16(a, b3, acc[3][t]);
    }
  }
  for (int k0 = 0; k0 < HID_DIM; k0 += 32) {  // h part: i,o,u, W cols [512,1024)
    v16bf a = load_a_bf16(ah, k0, lane);
#pragma unroll
    for (int t = 0; t < 4; ++t) {
      int nb = nbase + t * 16;
      v16bf b0 = load_b(w[0], nb, 512, k0, lane);
      v16bf b1 = load_b(w[1], nb, 512, k0, lane);
      v16bf b2 = load_b(w[2], nb, 512, k0, lane);
      acc[0][t] = wmma_bf16(a, b0, acc[0][t]);
      acc[1][t] = wmma_bf16(a, b1, acc[1][t]);
      acc[2][t] = wmma_bf16(a, b2, acc[2][t]);
    }
  }
  int mhi = (lane >> 4) << 3;
#pragma unroll
  for (int t = 0; t < 4; ++t) {
    int ncol = nbase + t * 16 + (lane & 15);
#pragma unroll
    for (int j = 0; j < 8; ++j) {
      int m = mt * 16 + mhi + j;
      if (m < n) {
        size_t idx = (size_t)m * HID_DIM + ncol;
        pre_i[idx] = acc[0][t][j];
        pre_o[idx] = acc[1][t][j];
        pre_u[idx] = acc[2][t][j];
        pre_fx[idx] = acc[3][t][j];
      }
    }
  }
}

// ---------------------------------------------------------------------------
// Forget gates. Block = 8 waves = 8 parents, one colgroup; wave = the 16
// children of its parent (one M-tile) x 64 cols. Wfh k-step tiles (4KB)
// double-buffered in LDS via async-to-LDS. Uniform control flow: out-of-range
// parents clamp and skip stores but still stage/barrier.
// f_k = sigmoid(pre_fx[p] + C_k@Wfh^T + bf); fC[p] = sum_k f_k * C_k.
// ---------------------------------------------------------------------------
__global__ __launch_bounds__(256) void k_forget(
    int n, const float* __restrict__ Cchild, const float* __restrict__ pre_fx,
    const float* __restrict__ bfb, const __bf16* __restrict__ wf,
    float* __restrict__ fC) {
  __shared__ __bf16 smem[2][64 * 32];  // 2 x 4KB
  int tid = threadIdx.x;
  int lane = tid & 31;
  int pg = blockIdx.x >> 3;
  int cg = blockIdx.x & 7;
  int p = pg * 8 + (tid >> 5);
  int pc = p < n ? p : n - 1;
  bool wr = p < n;
  int nbase = cg * 64;
  const float* arow = Cchild + (size_t)(pc * 16 + (lane & 15)) * HID_DIM;

  auto stage = [&](int buf, int k0) {
    int nn = tid >> 2, q = tid & 3;  // 256 chunks of 16B
    const __bf16* src = wf + (size_t)(nbase + nn) * WROW + 512 + k0 + q * 8;
    unsigned loff = (unsigned)(size_t)(&smem[buf][nn * 32 + q * 8]);
    async_ld16(loff, src);
  };

  stage(0, 0);
  v8f acc[4] = {};
  for (int ks = 0; ks < 16; ++ks) {
    int k0 = ks * 32;
    int cur = ks & 1;
    __syncthreads();
    if (ks + 1 < 16) {
      stage(cur ^ 1, k0 + 32);
      wait_async1();
    } else {
      wait_async0();
    }
    __syncthreads();
    v16bf a = load_a_f32(arow, k0, lane);
    const __bf16* bb = &smem[cur][0];
#pragma unroll
    for (int t = 0; t < 4; ++t)
      acc[t] = wmma_bf16(a, load_b_lds(bb, 0, t, lane), acc[t]);
  }

  int mhi = (lane >> 4) << 3;
#pragma unroll
  for (int t = 0; t < 4; ++t) {
    int ncol = nbase + t * 16 + (lane & 15);
    float fx = pre_fx[(size_t)pc * HID_DIM + ncol] + bfb[ncol];
    float part = 0.f;
#pragma unroll
    for (int j = 0; j < 8; ++j) {
      int m = mhi + j;
      float f = sigmoidf_(acc[t][j] + fx);
      part += f * Cchild[(size_t)(pc * 16 + m) * HID_DIM + ncol];
    }
    part += __shfl_down(part, 16, 32);
    if (wr && lane < 16) fC[(size_t)p * HID_DIM + ncol] = part;
  }
}

// ---------------------------------------------------------------------------
// Combine: C = sig(pre_i+bi)*tanh(pre_u+bu) + fC; H = sig(pre_o+bo)*tanh(C).
// One block = 16 sibling nodes x 512 cols; writes parent C and the NEXT
// level's h_sum directly. Level 0 writes d_out = [H0 | C0].
// ---------------------------------------------------------------------------
__global__ __launch_bounds__(512) void k_combine(
    int n, const float* __restrict__ pre_i, const float* __restrict__ pre_o,
    const float* __restrict__ pre_u, const float* __restrict__ fC,
    const float* __restrict__ bi, const float* __restrict__ bo,
    const float* __restrict__ bu, float* __restrict__ Cout,
    __bf16* __restrict__ hsum, float* __restrict__ dout, int final_level) {
  int c = threadIdx.x;  // 0..511
  int g = blockIdx.x;
  float bi_v = bi[c], bo_v = bo[c], bu_v = bu[c];
  float hs = 0.f;
  for (int mm = 0; mm < 16; ++mm) {
    int m = g * 16 + mm;
    if (m >= n) break;  // n is uniform
    size_t idx = (size_t)m * HID_DIM + c;
    float iv = sigmoidf_(pre_i[idx] + bi_v);
    float ov = sigmoidf_(pre_o[idx] + bo_v);
    float uv = tanhf(pre_u[idx] + bu_v);
    float C = iv * uv + fC[idx];
    float H = ov * tanhf(C);
    if (final_level) {
      dout[c] = H;
      dout[HID_DIM + c] = C;
    } else {
      Cout[idx] = C;
      hs += H;
    }
  }
  if (!final_level) hsum[(size_t)g * HID_DIM + c] = (__bf16)hs;
}

// ---------------------------------------------------------------------------
// Host side
// ---------------------------------------------------------------------------
extern "C" void kernel_launch(void* const* d_in, const int* in_sizes, int n_in,
                              void* d_out, int out_size, void* d_ws,
                              size_t ws_size, hipStream_t stream) {
  (void)in_sizes; (void)n_in; (void)out_size; (void)ws_size;
  const float* x    = (const float*)d_in[0];
  const float* wi_w = (const float*)d_in[1];
  const float* wi_b = (const float*)d_in[2];
  const float* wf_w = (const float*)d_in[3];
  const float* wf_b = (const float*)d_in[4];
  const float* wo_w = (const float*)d_in[5];
  const float* wo_b = (const float*)d_in[6];
  const float* wu_w = (const float*)d_in[7];
  const float* wu_b = (const float*)d_in[8];

  char* ws = (char*)d_ws;
  size_t off = 0;
  auto take = [&](size_t bytes) {
    void* p = ws + off;
    off = (off + bytes + 255) & ~(size_t)255;
    return p;
  };
  __bf16* wbf   = (__bf16*)take(4ull * WELEM * 2);           //  4.0 MB
  __bf16* xbf   = (__bf16*)take((size_t)NNODES * IN_DIM * 2);//  71.6 MB
  float*  Cleaf = (float*) take(65536ull * HID_DIM * 4);     // 128 MB
  float*  Csa   = (float*) take(4096ull * HID_DIM * 4);      //   8 MB
  float*  Csb   = (float*) take(4096ull * HID_DIM * 4);      //   8 MB
  __bf16* hsum  = (__bf16*)take(4096ull * HID_DIM * 2);      //   4 MB
  float*  pre_i = (float*) take(4096ull * HID_DIM * 4);      //   8 MB
  float*  pre_o = (float*) take(4096ull * HID_DIM * 4);
  float*  pre_u = (float*) take(4096ull * HID_DIM * 4);
  float*  pre_f = (float*) take(4096ull * HID_DIM * 4);
  float*  fC    = (float*) take(4096ull * HID_DIM * 4);      // total ~264 MB

  k_cast<<<2048, 256, 0, stream>>>(wi_w, wo_w, wu_w, wf_w, x, wbf, xbf);

  // Leaves: 512 parent-groups (8 parents each) x 8 colgroups.
  k_leaf<<<4096, 256, 0, stream>>>(xbf, wbf, wi_b, wo_b, wu_b, Cleaf, hsum);

  const int   ns[4]    = {4096, 256, 16, 1};
  const int   xoffs[4] = {273, 17, 1, 0};
  const float* Cch[4]  = {Cleaf, Csa, Csb, Csa};
  float*       Cout[4] = {Csa, Csb, Csa, Csb};  // level-0 Cout unused

  for (int li = 0; li < 4; ++li) {
    int n = ns[li];
    int mtiles = (n + 15) / 16;
    int gwaves = mtiles * 8;
    k_gates<<<(gwaves + 3) / 4, 128, 0, stream>>>(n, xoffs[li], xbf, hsum, wbf,
                                                  pre_i, pre_o, pre_u, pre_f);
    int fblocks = ((n + 7) / 8) * 8;  // parent-groups x 8 colgroups
    k_forget<<<fblocks, 256, 0, stream>>>(n, Cch[li], pre_f, wf_b,
                                          wbf + 3ull * WELEM, fC);
    k_combine<<<mtiles, 512, 0, stream>>>(n, pre_i, pre_o, pre_u, fC, wi_b,
                                          wo_b, wu_b, Cout[li], hsum,
                                          (float*)d_out, n == 1 ? 1 : 0);
  }
}
